// DAB_83545703842288
// MI455X (gfx1250) — matrix operand order repaired and checked by reading, not compiled, not run on hardware
//
#include <hip/hip_runtime.h>
#include <hip/hip_bf16.h>
#include <math.h>

typedef _Float16 h16;
typedef __attribute__((ext_vector_type(16))) _Float16 v16h;
typedef __attribute__((ext_vector_type(8)))  _Float16 v8h;
typedef __attribute__((ext_vector_type(8)))  float    v8f;

#define BB 4
#define CC 64
#define HH 160
#define WW 160
#define GG 8
#define OC_OFF 18
#define KSTEPS 18          // K = 576 = 18 steps of 32, tap-major: k = tap*64 + ci
#define XT_STRIDE 72       // padded ci stride (multiple of 8 halfs = 16B aligned)
#define PXT 32             // pixels per block (two 16-wide N tiles per wave)
#define XT_DX (PXT + 2)    // 34 halo columns

union F16x16 { v16h v; v8h h2[2]; h16 e[16]; };
union CFrag  { v8f  v; float e[8]; };

// ---------------------------------------------------------------------------
// Kernel 0: pack conv weights (fp32 -> f16) into WMMA A-fragment order.
//   flat idx = ((ks*NCO + co)*2 + hi)*16 + h
//   element h of the v16h holds k_local = h + hi*16  (contiguous k order)
//   global k = ks*32 + k_local ; tap = k/64 ; ci = k%64  (tap-major K)
// Offset table padded to 32 rows (rows 18..31 = 0).
// ---------------------------------------------------------------------------
__global__ __launch_bounds__(256)
void prep_wfrag_kernel(const float* __restrict__ ow,
                       const float* __restrict__ cw,
                       h16* __restrict__ wfrag_off,   // 18*32*2*16 = 18432
                       h16* __restrict__ wfrag_cv) {  // 18*64*2*16 = 36864
  const int i = blockIdx.x * 256 + threadIdx.x;
  if (i < 18432) {
    const int h  = i & 15;
    const int hi = (i >> 4) & 1;
    const int co = (i >> 5) & 31;
    const int ks = i >> 10;
    const int v2 = h >> 1, cb = h & 1;
    const int ka = ks * 32 + 2 * v2 + cb + hi * 16;
    const int tap = ka >> 6, ci = ka & 63;
    float val = (co < OC_OFF) ? ow[(co * CC + ci) * 9 + tap] : 0.f;
    wfrag_off[i] = (h16)val;
  } else if (i < 18432 + 36864) {
    const int j  = i - 18432;
    const int h  = j & 15;
    const int hi = (j >> 4) & 1;
    const int co = (j >> 5) & 63;
    const int ks = j >> 11;
    const int v2 = h >> 1, cb = h & 1;
    const int ka = ks * 32 + 2 * v2 + cb + hi * 16;
    const int tap = ka >> 6, ci = ka & 63;
    wfrag_cv[j] = (h16)cw[(co * CC + ci) * 9 + tap];
  }
}

// ---------------------------------------------------------------------------
// Conv 3x3 pad 1 as implicit GEMM with WMMA f32<-f16.  NCO = padded output
// channels (32 or 64); NCO/16 waves; each wave owns a 16(co) x 32(px) tile
// (two 16x16 WMMA N-tiles sharing one A fragment per k-step).
// x halo tile in LDS laid out [ky][dx][ci] so each B fragment is one
// contiguous 32-byte LDS read; A fragments pre-packed in global memory.
// ---------------------------------------------------------------------------
template <int NCO, bool HAS_ACT>
__global__ __launch_bounds__(NCO * 2)
void conv_wmma_kernel(const float* __restrict__ x,
                      const h16* __restrict__ wfrag,
                      const float* __restrict__ bias,
                      const float* __restrict__ pa,
                      float* __restrict__ outp,
                      int outChannels) {
  __shared__ h16 xt[3 * XT_DX * XT_STRIDE];  // [ky 0..2][dx 0..33][ci 0..63(+pad)]
  const int tid  = threadIdx.x;
  const int lane = tid & 31;
  const int wv   = tid >> 5;

  const int tilesPerRow = WW / PXT;                // 5
  const int t   = blockIdx.x;
  const int row = t / tilesPerRow;                 // 0..B*H-1
  const int x0  = (t % tilesPerRow) * PXT;
  const int b   = row / HH;
  const int y   = row % HH;

  for (int i = tid; i < CC * 3 * XT_DX; i += NCO * 2) {
    int ci = i / (3 * XT_DX); int r = i % (3 * XT_DX);
    int dy = r / XT_DX; int dx = r % XT_DX;
    int gy = y + dy - 1, gx = x0 + dx - 1;
    float v = 0.f;
    if (gy >= 0 && gy < HH && gx >= 0 && gx < WW)
      v = x[((size_t)(b * CC + ci) * HH + gy) * WW + gx];
    xt[(dy * XT_DX + dx) * XT_STRIDE + ci] = (h16)v;
  }
  __syncthreads();

  const int coBase = wv * 16;
  const int m  = lane & 15;
  const int hi = lane >> 4;                        // 0 or 1

  // Per-lane bases; all per-kstep offsets below are compile-time constants.
  const h16* aPtr = wfrag + ((size_t)(coBase + m) * 2 + hi) * 16;
  const h16* bPtr = xt + m * XT_STRIDE + hi * 16;

  CFrag c0, c1;
  #pragma unroll
  for (int r = 0; r < 8; ++r) { c0.e[r] = 0.f; c1.e[r] = 0.f; }

  #pragma unroll
  for (int ks = 0; ks < KSTEPS; ++ks) {
    const int tap = ks >> 1;                       // constant per step
    const int ky  = tap / 3, kx = tap % 3;
    const int ci0 = (ks & 1) * 32;

    F16x16 a, bf0, bf1;
    const h16* ap = aPtr + (size_t)ks * NCO * 32;
    a.h2[0] = *(const v8h*)(ap);
    a.h2[1] = *(const v8h*)(ap + 8);
    const h16* bp = bPtr + (ky * XT_DX + kx) * XT_STRIDE + ci0;
    bf0.h2[0] = *(const v8h*)(bp);
    bf0.h2[1] = *(const v8h*)(bp + 8);
    const h16* bq = bp + 16 * XT_STRIDE;           // second pixel tile
    bf1.h2[0] = *(const v8h*)(bq);
    bf1.h2[1] = *(const v8h*)(bq + 8);

    c0.v = __builtin_amdgcn_wmma_f32_16x16x32_f16(false, a.v, false, bf0.v,
                                                  (short)0, c0.v, false, false);
    c1.v = __builtin_amdgcn_wmma_f32_16x16x32_f16(false, a.v, false, bf1.v,
                                                  (short)0, c1.v, false, false);
  }

  const float alpha = HAS_ACT ? pa[0] : 0.f;
  #pragma unroll
  for (int r = 0; r < 8; ++r) {
    const int cod = coBase + (hi ? r + 8 : r);
    const size_t base = ((size_t)(b * outChannels + cod) * HH + y) * WW + x0 + m;
    if (!HAS_ACT) {
      outp[base]      = c0.e[r];
      outp[base + 16] = c1.e[r];
    } else if (cod < outChannels) {
      float v0 = c0.e[r] + bias[cod];
      float v1 = c1.e[r] + bias[cod];
      v0 = (v0 >= 0.f) ? v0 : alpha * v0;
      v1 = (v1 >= 0.f) ? v1 : alpha * v1;
      outp[base]      = v0;
      outp[base + 16] = v1;
    }
  }
}

// ---------------------------------------------------------------------------
// Deformable conv (deform_groups=1, groups=8).  One thread per (pixel, group):
// 8 in-ch x 9 taps bilinear gathers, 8x8x9 MACs.  Gather-bound -> VALU path.
// ---------------------------------------------------------------------------
__device__ __forceinline__ float dc_sample(const float* __restrict__ xc,
                                           int yy, int xx2) {
  return (yy >= 0 && yy < HH && xx2 >= 0 && xx2 < WW) ? xc[yy * WW + xx2] : 0.f;
}

__global__ __launch_bounds__(256)
void deform_conv_kernel(const float* __restrict__ x,
                        const float* __restrict__ dw,
                        const float* __restrict__ db,
                        const float* __restrict__ off_ws,
                        float* __restrict__ feat) {
  __shared__ float wl[CC * 8 * 9];   // [co][ci_in_group][tap]
  __shared__ float bl[CC];
  const int tid = threadIdx.x;
  for (int i = tid; i < CC * 8 * 9; i += 256) wl[i] = dw[i];
  if (tid < CC) bl[tid] = db[tid];
  __syncthreads();

  const int gid = blockIdx.x * 256 + tid;
  const int g   = gid & 7;
  const int pix = gid >> 3;
  const int b   = pix / (HH * WW);
  const int rem = pix - b * HH * WW;
  const int y   = rem / WW;
  const int xx  = rem - y * WW;

  float acc[8];
  #pragma unroll
  for (int o = 0; o < 8; ++o) acc[o] = bl[g * 8 + o];

  const float* xb   = x + (size_t)b * CC * HH * WW;
  const float* offb = off_ws + (size_t)b * OC_OFF * HH * WW;

  #pragma unroll
  for (int tap = 0; tap < 9; ++tap) {
    const int ky = tap / 3, kx = tap % 3;
    const float dy = offb[((tap * 2 + 0) * HH + y) * WW + xx];
    const float dx = offb[((tap * 2 + 1) * HH + y) * WW + xx];
    const float py = (float)(y - 1 + ky) + dy;
    const float px = (float)(xx - 1 + kx) + dx;
    const float y0f = floorf(py), x0f = floorf(px);
    const float ly = py - y0f, lx = px - x0f;
    const int y0 = (int)y0f, xq = (int)x0f;
    const float w00 = (1.f - ly) * (1.f - lx);
    const float w01 = (1.f - ly) * lx;
    const float w10 = ly * (1.f - lx);
    const float w11 = ly * lx;
    #pragma unroll
    for (int cin = 0; cin < 8; ++cin) {
      const float* xc = xb + (size_t)(g * 8 + cin) * HH * WW;
      float v = w00 * dc_sample(xc, y0,     xq)
              + w01 * dc_sample(xc, y0,     xq + 1)
              + w10 * dc_sample(xc, y0 + 1, xq)
              + w11 * dc_sample(xc, y0 + 1, xq + 1);
      #pragma unroll
      for (int o = 0; o < 8; ++o)
        acc[o] += wl[((g * 8 + o) * 8 + cin) * 9 + tap] * v;
    }
  }
  #pragma unroll
  for (int o = 0; o < 8; ++o)
    feat[((size_t)(b * CC + g * 8 + o) * HH + y) * WW + xx] = acc[o];
}

// ---------------------------------------------------------------------------
// Per-row (W=160) softmax of logits (in-place in d_out), then
// out = softmax * feat + x.  One block (160 threads) per (b,c,h) row.
// ---------------------------------------------------------------------------
__global__ __launch_bounds__(WW)
void softmax_combine_kernel(const float* __restrict__ x,
                            const float* __restrict__ feat,
                            float* __restrict__ out) {
  __shared__ float s[WW];
  const int j = threadIdx.x;
  const size_t base = (size_t)blockIdx.x * WW;

  const float v = out[base + j];
  s[j] = v;
  __syncthreads();
  for (int st = 128; st >= 1; st >>= 1) {
    if (j < st && j + st < WW) s[j] = fmaxf(s[j], s[j + st]);
    __syncthreads();
  }
  const float mx = s[0];
  __syncthreads();

  const float e = __expf(v - mx);
  s[j] = e;
  __syncthreads();
  for (int st = 128; st >= 1; st >>= 1) {
    if (j < st && j + st < WW) s[j] += s[j + st];
    __syncthreads();
  }
  const float inv = 1.f / s[0];

  out[base + j] = e * inv * feat[base + j] + x[base + j];
}

// ---------------------------------------------------------------------------
extern "C" void kernel_launch(void* const* d_in, const int* in_sizes, int n_in,
                              void* d_out, int out_size, void* d_ws, size_t ws_size,
                              hipStream_t stream) {
  (void)in_sizes; (void)n_in; (void)out_size; (void)ws_size;
  const float* x  = (const float*)d_in[0];   // [4,64,160,160]
  const float* ow = (const float*)d_in[1];   // [18,64,3,3]
  const float* ob = (const float*)d_in[2];   // [18]
  const float* pa = (const float*)d_in[3];   // [1]
  const float* dw = (const float*)d_in[4];   // [64,8,3,3]
  const float* db = (const float*)d_in[5];   // [64]
  const float* cw = (const float*)d_in[6];   // [64,64,3,3]
  float* out = (float*)d_out;

  float* off_ws  = (float*)d_ws;                                  // 4*18*160*160 f32
  float* feat_ws = off_ws + (size_t)BB * OC_OFF * HH * WW;        // 4*64*160*160 f32
  h16*   wfo     = (h16*)(feat_ws + (size_t)BB * CC * HH * WW);   // 18432 h16
  h16*   wfc     = wfo + 18432;                                   // 36864 h16

  const int pixTiles = BB * HH * (WW / PXT);                      // 3200

  prep_wfrag_kernel<<<(18432 + 36864 + 255) / 256, 256, 0, stream>>>(ow, cw, wfo, wfc);
  conv_wmma_kernel<32, true><<<pixTiles, 64, 0, stream>>>(x, wfo, ob, pa, off_ws, OC_OFF);
  deform_conv_kernel<<<(BB * HH * WW * GG) / 256, 256, 0, stream>>>(
      x, dw, db, off_ws, feat_ws);
  conv_wmma_kernel<64, false><<<pixTiles, 128, 0, stream>>>(x, wfc, nullptr, nullptr, out, CC);
  softmax_combine_kernel<<<BB * CC * HH, WW, 0, stream>>>(x, feat_ws, out);
}
